// ModalityUntiedFeedForward_55336358642914
// MI455X (gfx1250) — matrix operand order repaired and checked by reading, not compiled, not run on hardware
//
#include <hip/hip_runtime.h>
#include <hip/hip_bf16.h>

#define DIMX 1024
#define HIDX 2816
#define NEXP 2
#define NTOK 32768
#define HT   256                 // hidden chunk per iteration
#define XSst (DIMX + 8)          // LDS stride (halves), +8 kills bank conflicts
#define HSst (HT + 8)

typedef _Float16 h4  __attribute__((ext_vector_type(4)));
typedef _Float16 h8  __attribute__((ext_vector_type(8)));
typedef _Float16 h16 __attribute__((ext_vector_type(16)));
typedef float    f8  __attribute__((ext_vector_type(8)));

__device__ __forceinline__ f8 wmma_f16(h16 a, h16 b, f8 c) {
  // D = A(16x32 f16) x B(32x16 f16) + C(16x16 f32)
  return __builtin_amdgcn_wmma_f32_16x16x32_f16(false, a, false, b, (short)0, c,
                                                false, false);
}

// ---------------------------------------------------------------- routing ---
__global__ void zero_cnt_kernel(int* cnt) {
  if (threadIdx.x < NEXP) cnt[threadIdx.x] = 0;
}

__global__ void compact_kernel(const int* __restrict__ masks,
                               int* __restrict__ cnt, int* __restrict__ idxl) {
  int t = blockIdx.x * blockDim.x + threadIdx.x;
  if (t >= NTOK) return;
  int e = masks[t] ? 0 : (masks[NTOK + t] ? 1 : -1);  // modality 0 wins
  if (e >= 0) {
    int p = atomicAdd(&cnt[e], 1);
    idxl[e * NTOK + p] = t;
  }
}

// ------------------------------------------------- weight fragment packing --
// F layout: ((nt*KT + kt)*32 + lane)*16 + j  with
//   n = nt*16 + lane%16 ,  k = kt*32 + (j/8)*16 + (lane/16)*8 + j%8
// == per-lane B-fragment order of v_wmma_f32_16x16x32_f16 (ISA 7.12.2).
__global__ void pack_w_kernel(const float* __restrict__ W, _Float16* __restrict__ F,
                              int inc, int ktn, long total) {
  long i = (long)blockIdx.x * blockDim.x + threadIdx.x;
  if (i >= total) return;
  int  j    = (int)(i & 15);
  int  lane = (int)((i >> 4) & 31);
  long t    = i >> 9;
  int  kt   = (int)(t % ktn);
  int  nt   = (int)(t / ktn);
  int  n = nt * 16 + (lane & 15);
  int  k = kt * 32 + (j >> 3) * 16 + (lane >> 4) * 8 + (j & 7);
  F[i] = (_Float16)W[(long)n * inc + k];
}

// ------------------------------------------------------------- fused FFN ----
__global__ __launch_bounds__(512)
void moe_ffn_kernel(const float* __restrict__ x, const float* __restrict__ norm_w,
                    const int* __restrict__ cnt, const int* __restrict__ idxl,
                    const _Float16* __restrict__ w1f, const _Float16* __restrict__ w3f,
                    const _Float16* __restrict__ w2f, float* __restrict__ out) {
  __shared__ _Float16 xs[32 * XSst];   // x tile, fp16
  __shared__ _Float16 hs[32 * HSst];   // h chunk, fp16
  __shared__ float    part[32 * 128];  // RMSNorm partials
  __shared__ float    rsv[32];
  __shared__ int      tok[32];

  const int bx   = blockIdx.x;
  const int e    = bx >> 10;           // 1024 token-tiles per expert
  const int tile = bx & 1023;
  const int cntE = cnt[e];
  if (tile * 32 >= cntE) return;       // uniform early-out, before any barrier
  const int valid = min(32, cntE - tile * 32);

  const int tid = threadIdx.x;
  if (tid < 32)
    tok[tid] = idxl[e * NTOK + tile * 32 + min(tid, valid - 1)];
  __syncthreads();

  // ---- stage x tile into LDS as fp16 (coalesced float4 reads) ----
  for (int i = 0; i < 16; ++i) {
    int fid = tid + i * 512;                 // 8192 float4 total
    int row = fid >> 8, c4 = fid & 255;
    unsigned xoff = (unsigned)tok[row] * DIMX + (unsigned)(c4 * 4);
    float4 v = *(const float4*)(x + xoff);   // saddr + 32-bit voffset form
    h4 hv; hv[0] = (_Float16)v.x; hv[1] = (_Float16)v.y;
           hv[2] = (_Float16)v.z; hv[3] = (_Float16)v.w;
    *(h4*)(&xs[row * XSst + c4 * 4]) = hv;
  }
  __syncthreads();

  const int w = tid >> 5, lane = tid & 31;
  const int mt = w >> 3, g = w & 7;          // mt: token half, g: column group
  const int lo = lane & 15, hiH = lane >> 4;

  const unsigned ebase12 = (unsigned)e * (HIDX * DIMX);  // halves, < 2^23
  const unsigned lane16  = (unsigned)lane * 16u;

  const f8 zf = {0.f, 0.f, 0.f, 0.f, 0.f, 0.f, 0.f, 0.f};
  f8 accO[8];
#pragma unroll
  for (int j = 0; j < 8; ++j) accO[j] = zf;

  const _Float16* xrow = &xs[(mt * 16 + lo) * XSst];
  const _Float16* hrow = &hs[(mt * 16 + lo) * HSst];

  for (int hc = 0; hc < HIDX / HT; ++hc) {   // 11 chunks
    // ---- (a) h = silu(x@W1^T) * (x@W3^T), 2 column tiles per wave ----
    f8 a1[2] = {zf, zf}, a3[2] = {zf, zf};
#pragma unroll 4
    for (int kt = 0; kt < DIMX / 32; ++kt) {
      h8 l0 = *(const h8*)(xrow + kt * 32 + hiH * 8);
      h8 l1 = *(const h8*)(xrow + kt * 32 + hiH * 8 + 16);
      h16 A = __builtin_shufflevector(l0, l1, 0, 1, 2, 3, 4, 5, 6, 7,
                                      8, 9, 10, 11, 12, 13, 14, 15);
#pragma unroll
      for (int t = 0; t < 2; ++t) {
        int      NTg = hc * 16 + g * 2 + t;           // global HID tile
        unsigned off = ebase12 +
                       ((unsigned)(NTg * 32 + kt) * 32u) * 16u + lane16;
        h16 B1 = *(const h16*)(w1f + off);
        h16 B3 = *(const h16*)(w3f + off);
        a1[t] = wmma_f16(A, B1, a1[t]);
        a3[t] = wmma_f16(A, B3, a3[t]);
      }
    }
#pragma unroll
    for (int t = 0; t < 2; ++t) {
      int col = (g * 2 + t) * 16 + lo;
#pragma unroll
      for (int r = 0; r < 8; ++r) {
        float aa = a1[t][r];
        float hv = (aa / (1.0f + __expf(-aa))) * a3[t][r];
        hs[(mt * 16 + r + 8 * hiH) * HSst + col] = (_Float16)hv;
      }
    }
    __syncthreads();

    // ---- (b) o += h_chunk @ W2^T, 8 persistent output tiles per wave ----
#pragma unroll 2
    for (int kt2 = 0; kt2 < HT / 32; ++kt2) {
      h8 l0 = *(const h8*)(hrow + kt2 * 32 + hiH * 8);
      h8 l1 = *(const h8*)(hrow + kt2 * 32 + hiH * 8 + 16);
      h16 A = __builtin_shufflevector(l0, l1, 0, 1, 2, 3, 4, 5, 6, 7,
                                      8, 9, 10, 11, 12, 13, 14, 15);
      int ktg = hc * (HT / 32) + kt2;               // global K tile in HID
#pragma unroll
      for (int j = 0; j < 8; ++j) {
        int      nt  = g * 8 + j;
        unsigned off = ebase12 +
                       ((unsigned)(nt * (HIDX / 32) + ktg) * 32u) * 16u + lane16;
        h16 B2 = *(const h16*)(w2f + off);
        accO[j] = wmma_f16(A, B2, accO[j]);
      }
    }
    __syncthreads();                               // protect hs for next chunk
  }

  // ---- RMSNorm (fixed-order reduction => deterministic) ----
#pragma unroll
  for (int r = 0; r < 8; ++r) {
    float s = 0.f;
#pragma unroll
    for (int j = 0; j < 8; ++j) { float v = accO[j][r]; s += v * v; }
    part[(mt * 16 + r + 8 * hiH) * 128 + g * 16 + lo] = s;
  }
  __syncthreads();
  if (tid < 32) {
    float s = 0.f;
    for (int c = 0; c < 128; ++c) s += part[tid * 128 + c];
    rsv[tid] = rsqrtf(s * (1.0f / DIMX) + 1e-5f);
  }
  __syncthreads();

  float gw[8];
#pragma unroll
  for (int j = 0; j < 8; ++j)
    gw[j] = norm_w[(unsigned)(e * DIMX + (g * 8 + j) * 16 + lo)];
#pragma unroll
  for (int r = 0; r < 8; ++r) {
    int R = mt * 16 + r + 8 * hiH;
    if (tile * 32 + R < cntE) {
      float    sc = rsv[R];
      unsigned ob = (unsigned)tok[R] * DIMX;       // < 2^25
#pragma unroll
      for (int j = 0; j < 8; ++j)
        out[ob + (unsigned)((g * 8 + j) * 16 + lo)] = accO[j][r] * sc * gw[j];
    }
  }
}

// ----------------------------------------------------------------- launch ---
extern "C" void kernel_launch(void* const* d_in, const int* in_sizes, int n_in,
                              void* d_out, int out_size, void* d_ws, size_t ws_size,
                              hipStream_t stream) {
  const float* x     = (const float*)d_in[0];
  const int*   masks = (const int*)d_in[1];
  const float* w1    = (const float*)d_in[2];
  const float* w3    = (const float*)d_in[3];
  const float* w2    = (const float*)d_in[4];
  const float* nw    = (const float*)d_in[5];
  float*       out   = (float*)d_out;

  char* ws   = (char*)d_ws;
  int*  cnt  = (int*)ws;                                  // [2]
  int*  idxl = (int*)(ws + 256);                          // [2][NTOK]
  const long mhalves = (long)NEXP * HIDX * DIMX;          // per weight tensor
  _Float16* w1f = (_Float16*)(ws + 262656);
  _Float16* w3f = w1f + mhalves;
  _Float16* w2f = w3f + mhalves;

  zero_cnt_kernel<<<1, 64, 0, stream>>>(cnt);
  compact_kernel<<<(NTOK + 255) / 256, 256, 0, stream>>>(masks, cnt, idxl);

  int blocks = (int)((mhalves + 255) / 256);
  pack_w_kernel<<<blocks, 256, 0, stream>>>(w1, w1f, DIMX, DIMX / 32, mhalves);
  pack_w_kernel<<<blocks, 256, 0, stream>>>(w3, w3f, DIMX, DIMX / 32, mhalves);
  pack_w_kernel<<<blocks, 256, 0, stream>>>(w2, w2f, HIDX, HIDX / 32, mhalves);

  moe_ffn_kernel<<<NEXP * (NTOK / 32), 512, 0, stream>>>(
      x, nw, cnt, idxl, w1f, w3f, w2f, out);
}
